// KG_ecosys_18640158065240
// MI455X (gfx1250) — matrix operand order repaired and checked by reading, not compiled, not run on hardware
//
#include <hip/hip_runtime.h>
#include <hip/hip_bf16.h>
#include <stdint.h>

typedef __attribute__((ext_vector_type(16))) _Float16 v16h;
typedef __attribute__((ext_vector_type(8)))  _Float16 v8h;
typedef __attribute__((ext_vector_type(8)))  float    v8f;

#define DEV __device__ __forceinline__

constexpr int B = 4096, T = 365, D = 21, H = 64;
constexpr int G = 3 * H;            // 192 gate rows
constexpr int WAVES = 4;            // waves per block, each owns a 16-row batch tile

// ---------------- LDS layout ----------------
// f16 weight section (element offsets)
constexpr int E_WIH1 = 0;                    // 192 x 32 (K=21 padded)
constexpr int E_WHH1 = E_WIH1 + G * 32;      // 192 x 64
constexpr int E_WIH2 = E_WHH1 + G * 64;      // 192 x 32 (K=22 padded)
constexpr int E_WHH2 = E_WIH2 + G * 32;
constexpr int E_WIH3 = E_WHH2 + G * 64;      // 192 x 128
constexpr int E_WHH3 = E_WIH3 + G * 128;
constexpr int E_WIH4 = E_WHH3 + G * 64;      // 192 x 128
constexpr int E_WHH4 = E_WIH4 + G * 128;
constexpr int E_FW   = E_WHH4 + G * 64;      // 4 x (16 x 64), rows padded w/ zeros
constexpr int E_END  = E_FW + 4 * 16 * 64;   // 114688 f16
constexpr int BYTES_W = E_END * 2;           // 229376 B

// float section (element offsets from float base)
constexpr int F_BRZ = 0;                     // 4 x 128 : bih+bhh for r,z gates
constexpr int F_BIN = 4 * 128;               // 4 x 64  : bih for n gate
constexpr int F_BHN = F_BIN + 4 * 64;        // 4 x 64  : bhh for n gate
constexpr int F_FB  = F_BHN + 4 * 64;        // 4 x 16 (zero padded) fc biases
constexpr int F_SC  = F_FB + 64;             // scale[21] then shift[21]
constexpr int F_END = F_SC + 42;
constexpr int BYTES_F = F_END * 4;

constexpr int PW_BASE   = ((BYTES_W + BYTES_F + 127) / 128) * 128;
constexpr int PW_X      = 0;                 // 16x32 f16 x-tile (col21=LAI, 22..31=0)
constexpr int PW_H      = 1024;              // 4 x (16x64 f16) h buffers
constexpr int PW_LAI    = PW_H + 4 * 2048;   // 16 f32
constexpr int PW_STRIDE = 9344;
constexpr int SMEM_BYTES = PW_BASE + WAVES * PW_STRIDE;  // ~265 KB < 320 KB/WGP

// ---------------- WMMA helpers ----------------
DEV v8f wmma(v16h a, v16h b, v8f c) {
  return __builtin_amdgcn_wmma_f32_16x16x32_f16(false, a, false, b, (short)0, c, false, false);
}

// A-fragment (16x32 f16, M across lanes, K across VGPRs) from LDS row-major tile.
// lane<16: row=lane, K={kb..kb+7, kb+16..kb+23}; lane>=16: row=lane-16, K={kb+8.., kb+24..}
DEV v16h load_a(const _Float16* buf, int ldk, int kb, int lane) {
  int row = lane & 15;
  int k0  = kb + ((lane & 16) ? 8 : 0);
  const _Float16* p = buf + row * ldk + k0;
  v8h lo = *(const v8h*)(p);
  v8h hi = *(const v8h*)(p + 16);
  v16h r;
#pragma unroll
  for (int i = 0; i < 8; ++i) { r[i] = lo[i]; r[i + 8] = hi[i]; }
  return r;
}

// B-fragment (32x16 f16, N across lanes, K across VGPRs) from row-major W[n][k]:
// column n of B == row n of W -> K={kb..kb+7, kb+16..kb+23} (lane<16) or {kb+8.., kb+24..} (lane>=16).
DEV v16h load_b(const _Float16* w, int ldk, int ntile, int kb, int lane) {
  int n  = (ntile << 4) + (lane & 15);
  int k0 = kb + ((lane & 16) ? 8 : 0);
  const _Float16* p = w + n * ldk + k0;
  v8h lo = *(const v8h*)(p);
  v8h hi = *(const v8h*)(p + 16);
  v16h r;
#pragma unroll
  for (int i = 0; i < 8; ++i) { r[i] = lo[i]; r[i + 8] = hi[i]; }
  return r;
}

// Fast activations using hardware v_exp_f32 / v_rcp_f32 (TRANS ops that
// co-execute with the WMMA pipe) instead of IEEE v_div_scale sequences.
DEV float sigm(float x) {
  return __builtin_amdgcn_rcpf(1.0f + __expf(-x));
}
// tanh(x) = 1 - 2/(exp(2x)+1); exact at 0, saturates correctly at +/-inf.
DEV float tanh_fast(float x) {
  return 1.0f - 2.0f * __builtin_amdgcn_rcpf(__expf(2.0f * x) + 1.0f);
}

// One GRU cell update for a 16-row tile. Gate order (r,z,n): N-tiles j, j+4, j+8.
// Accumulators start at zero (WMMA SRC2 inline 0); biases folded into epilogue.
template <int NKI>
DEV void gru(const _Float16* wih, int ldki, const _Float16* whh,
             const float* brz, const float* bin, const float* bhn,
             const v16h (&ai)[NKI], const v16h (&ah)[2],
             v8f (&hm)[4], _Float16* hbuf, int ncol, int mo, int lane) {
#pragma unroll
  for (int j = 0; j < 4; ++j) {
    v8f ar = {}, az = {}, an = {};
#pragma unroll
    for (int k = 0; k < NKI; ++k) {
      ar = wmma(ai[k], load_b(wih, ldki, j,     k * 32, lane), ar);
      az = wmma(ai[k], load_b(wih, ldki, j + 4, k * 32, lane), az);
      an = wmma(ai[k], load_b(wih, ldki, j + 8, k * 32, lane), an);
    }
    v8f hr = {}, hz = {}, hn = {};
#pragma unroll
    for (int k = 0; k < 2; ++k) {
      hr = wmma(ah[k], load_b(whh, 64, j,     k * 32, lane), hr);
      hz = wmma(ah[k], load_b(whh, 64, j + 4, k * 32, lane), hz);
      hn = wmma(ah[k], load_b(whh, 64, j + 8, k * 32, lane), hn);
    }
    float brz_r = brz[j * 16 + ncol];
    float brz_z = brz[64 + j * 16 + ncol];
    float bn_i  = bin[j * 16 + ncol];
    float bn_h  = bhn[j * 16 + ncol];
    v8f hnew;
#pragma unroll
    for (int v = 0; v < 8; ++v) {
      float r = sigm(ar[v] + hr[v] + brz_r);
      float z = sigm(az[v] + hz[v] + brz_z);
      float n = tanh_fast(an[v] + bn_i + r * (hn[v] + bn_h));
      hnew[v] = (1.0f - z) * n + z * hm[j][v];
    }
    hm[j] = hnew;
#pragma unroll
    for (int v = 0; v < 8; ++v)
      hbuf[(v + mo) * 64 + j * 16 + ncol] = (_Float16)hnew[v];
  }
}

DEV v8f fc16(const _Float16* w, const v16h (&ah)[2], int lane) {
  v8f c = {};
  c = wmma(ah[0], load_b(w, 64, 0, 0, lane), c);
  c = wmma(ah[1], load_b(w, 64, 0, 32, lane), c);
  return c;
}

DEV void emit(v8f c, float bias, float* out, int odim, int base, int t, int ncol, int mo) {
  if (ncol < odim) {
#pragma unroll
    for (int v = 0; v < 8; ++v)
      out[((size_t)(base + v + mo) * T + t) * odim + ncol] = c[v] + bias;
  }
}

DEV void fill_w(_Float16* dst, const float* src, int rows, int K, int Kp, int tid, int nt) {
  for (int i = tid; i < rows * K; i += nt) {
    int r = i / K, c = i - r * K;
    dst[r * Kp + c] = (_Float16)src[i];
  }
}

struct Params {
  const float* x;
  const float* bn;   // ws+64: scale[21], shift[21]
  const float *wih1, *whh1, *bih1, *bhh1;
  const float *wih2, *whh2, *bih2, *bhh2;
  const float *wih3, *whh3, *bih3, *bhh3;
  const float *wih4, *whh4, *bih4, *bhh4;
  const float *fw1, *fb1, *fw2, *fb2, *fw3, *fb3, *fw4, *fb4;
  float *o1, *o2, *o3, *o4;
};

// ---------------- BN pre-pass ----------------
__global__ void bn_stats(const float* __restrict__ x, float* acc, unsigned N) {
  __shared__ float ss[21], sq[21];
  if (threadIdx.x < 21) { ss[threadIdx.x] = 0.f; sq[threadIdx.x] = 0.f; }
  __syncthreads();
  for (unsigned i = blockIdx.x * blockDim.x + threadIdx.x; i < N; i += gridDim.x * blockDim.x) {
    float v = x[i];
    unsigned c = i % 21u;
    atomicAdd(&ss[c], v);
    atomicAdd(&sq[c], v * v);
  }
  __syncthreads();
  if (threadIdx.x < 21) {
    atomicAdd(&acc[threadIdx.x], ss[threadIdx.x]);
    atomicAdd(&acc[21 + threadIdx.x], sq[threadIdx.x]);
  }
}

__global__ void bn_finalize(const float* acc, const float* gamma, const float* beta,
                            float* sc, int count) {
  int c = threadIdx.x;
  if (c < 21) {
    float mean = acc[c] / (float)count;
    float var  = acc[21 + c] / (float)count - mean * mean;
    float inv  = rsqrtf(var + 1e-5f);
    float s    = gamma[c] * inv;
    sc[c]      = s;
    sc[21 + c] = beta[c] - mean * s;
  }
}

// ---------------- main persistent GRU kernel ----------------
__global__ __launch_bounds__(32 * WAVES, 1) void ecosys_main(Params p) {
  extern __shared__ char smem[];
  _Float16* W  = (_Float16*)smem;
  float*    Fb = (float*)(smem + BYTES_W);

  int tid = threadIdx.x, nt = blockDim.x;

  // Zero all of LDS (weight padding, h buffers, x pad cols, lai).
  for (int i = tid; i < SMEM_BYTES / 4; i += nt) ((uint32_t*)smem)[i] = 0u;
  __syncthreads();

  // f32 -> f16 weight conversion into LDS (row-major, K padded)
  fill_w(W + E_WIH1, p.wih1, G, 21, 32, tid, nt);
  fill_w(W + E_WHH1, p.whh1, G, 64, 64, tid, nt);
  fill_w(W + E_WIH2, p.wih2, G, 22, 32, tid, nt);
  fill_w(W + E_WHH2, p.whh2, G, 64, 64, tid, nt);
  fill_w(W + E_WIH3, p.wih3, G, 128, 128, tid, nt);
  fill_w(W + E_WHH3, p.whh3, G, 64, 64, tid, nt);
  fill_w(W + E_WIH4, p.wih4, G, 128, 128, tid, nt);
  fill_w(W + E_WHH4, p.whh4, G, 64, 64, tid, nt);
  fill_w(W + E_FW + 0 * 1024, p.fw1, 1, 64, 64, tid, nt);
  fill_w(W + E_FW + 1 * 1024, p.fw2, 3, 64, 64, tid, nt);
  fill_w(W + E_FW + 2 * 1024, p.fw3, 3, 64, 64, tid, nt);
  fill_w(W + E_FW + 3 * 1024, p.fw4, 2, 64, 64, tid, nt);

  // Bias prep: r/z gates use combined bih+bhh; n gate keeps bih_n / bhh_n split.
  {
    const float* bihs[4] = { p.bih1, p.bih2, p.bih3, p.bih4 };
    const float* bhhs[4] = { p.bhh1, p.bhh2, p.bhh3, p.bhh4 };
#pragma unroll
    for (int g = 0; g < 4; ++g) {
      for (int i = tid; i < 128; i += nt)
        Fb[F_BRZ + g * 128 + i] = bihs[g][i] + bhhs[g][i];
      for (int i = tid; i < 64; i += nt) {
        Fb[F_BIN + g * 64 + i] = bihs[g][128 + i];
        Fb[F_BHN + g * 64 + i] = bhhs[g][128 + i];
      }
    }
  }
  if (tid == 0) {
    Fb[F_FB + 0] = p.fb1[0];
    for (int i = 0; i < 3; ++i) Fb[F_FB + 16 + i] = p.fb2[i];
    for (int i = 0; i < 3; ++i) Fb[F_FB + 32 + i] = p.fb3[i];
    for (int i = 0; i < 2; ++i) Fb[F_FB + 48 + i] = p.fb4[i];
  }
  if (tid < 42) Fb[F_SC + tid] = p.bn[tid];
  __syncthreads();

  // Per-wave persistent state: each wave owns a 16-row batch tile.
  int wave = tid >> 5, lane = tid & 31;
  int base = (blockIdx.x * WAVES + wave) * 16;
  char* pw = smem + PW_BASE + wave * PW_STRIDE;
  _Float16* xb  = (_Float16*)(pw + PW_X);
  _Float16* hb1 = (_Float16*)(pw + PW_H);
  _Float16* hb2 = hb1 + 1024;
  _Float16* hb3 = hb1 + 2048;
  _Float16* hb4 = hb1 + 3072;
  float* lai = (float*)(pw + PW_LAI);

  int ncol = lane & 15;
  int mo   = (lane & 16) ? 8 : 0;
  int lrow = lane >> 1;
  int half = lane & 1;
  const float* xrow = p.x + (size_t)(base + lrow) * T * D;
  const float* sc = Fb + F_SC;
  const float* sh = Fb + F_SC + 21;
  float fb1v = Fb[F_FB + ncol];
  float fb2v = Fb[F_FB + 16 + ncol];
  float fb3v = Fb[F_FB + 32 + ncol];
  float fb4v = Fb[F_FB + 48 + ncol];

  v8f h1m[4] = {}, h2m[4] = {}, h3m[4] = {}, h4m[4] = {};

  for (int t = 0; t < T; ++t) {
    // Build x tile in LDS: BN-affine f32->f16, col 21 = LAI feedback, 22..31 stay 0.
    const float* gp = xrow + (size_t)t * D;
    if (!half) {
#pragma unroll
      for (int d = 0; d < 11; ++d) xb[lrow * 32 + d] = (_Float16)(gp[d] * sc[d] + sh[d]);
    } else {
#pragma unroll
      for (int d = 11; d < 21; ++d) xb[lrow * 32 + d] = (_Float16)(gp[d] * sc[d] + sh[d]);
      xb[lrow * 32 + 21] = (_Float16)lai[lrow];
    }
    if (t + 4 < T) __builtin_prefetch(gp + 4 * D, 0, 1);

    v16h ax[1] = { load_a(xb, 32, 0, lane) };

    // GRU1 (x -> h1), read h1_old frags before overwrite
    v16h a1o[2] = { load_a(hb1, 64, 0, lane), load_a(hb1, 64, 32, lane) };
    gru<1>(W + E_WIH1, 32, W + E_WHH1,
           Fb + F_BRZ, Fb + F_BIN, Fb + F_BHN,
           ax, a1o, h1m, hb1, ncol, mo, lane);

    // GRU2 ([x, lai] -> h2): same x fragment, wih2 col 21 carries the LAI weight
    v16h a2o[2] = { load_a(hb2, 64, 0, lane), load_a(hb2, 64, 32, lane) };
    gru<1>(W + E_WIH2, 32, W + E_WHH2,
           Fb + F_BRZ + 128, Fb + F_BIN + 64, Fb + F_BHN + 64,
           ax, a2o, h2m, hb2, ncol, mo, lane);

    v16h a1n[2] = { load_a(hb1, 64, 0, lane), load_a(hb1, 64, 32, lane) };
    v16h a2n[2] = { load_a(hb2, 64, 0, lane), load_a(hb2, 64, 32, lane) };
    emit(fc16(W + E_FW + 0 * 1024, a1n, lane), fb1v, p.o1, 1, base, t, ncol, mo);
    emit(fc16(W + E_FW + 1 * 1024, a2n, lane), fb2v, p.o2, 3, base, t, ncol, mo);

    // GRU3 ([h1, h2] -> h3)
    v16h a3o[2] = { load_a(hb3, 64, 0, lane), load_a(hb3, 64, 32, lane) };
    v16h ai3[4] = { a1n[0], a1n[1], a2n[0], a2n[1] };
    gru<4>(W + E_WIH3, 128, W + E_WHH3,
           Fb + F_BRZ + 256, Fb + F_BIN + 128, Fb + F_BHN + 128,
           ai3, a3o, h3m, hb3, ncol, mo, lane);

    v16h a3n[2] = { load_a(hb3, 64, 0, lane), load_a(hb3, 64, 32, lane) };
    emit(fc16(W + E_FW + 2 * 1024, a3n, lane), fb3v, p.o3, 3, base, t, ncol, mo);

    // GRU4 ([h1, h3] -> h4)
    v16h a4o[2] = { load_a(hb4, 64, 0, lane), load_a(hb4, 64, 32, lane) };
    v16h ai4[4] = { a1n[0], a1n[1], a3n[0], a3n[1] };
    gru<4>(W + E_WIH4, 128, W + E_WHH4,
           Fb + F_BRZ + 384, Fb + F_BIN + 192, Fb + F_BHN + 192,
           ai4, a4o, h4m, hb4, ncol, mo, lane);

    v16h a4n[2] = { load_a(hb4, 64, 0, lane), load_a(hb4, 64, 32, lane) };
    v8f c4 = fc16(W + E_FW + 3 * 1024, a4n, lane);
    emit(c4, fb4v, p.o4, 2, base, t, ncol, mo);

    // LAI feedback (detached): col 0 of (c4 + bias) -> lai[16] for next step
    if (ncol == 0) {
#pragma unroll
      for (int v = 0; v < 8; ++v) lai[v + mo] = c4[v] + fb4v;
    }
  }
}

extern "C" void kernel_launch(void* const* d_in, const int* in_sizes, int n_in,
                              void* d_out, int out_size, void* d_ws, size_t ws_size,
                              hipStream_t stream) {
  (void)in_sizes; (void)n_in; (void)out_size; (void)ws_size;
  const float* x     = (const float*)d_in[0];
  const float* gamma = (const float*)d_in[1];
  const float* beta  = (const float*)d_in[2];
  float* wsf = (float*)d_ws;

  // BN stats -> scale/shift at wsf[64..105]
  hipMemsetAsync(d_ws, 0, 42 * sizeof(float), stream);
  unsigned N = (unsigned)B * T * D;
  bn_stats<<<512, 256, 0, stream>>>(x, wsf, N);
  bn_finalize<<<1, 32, 0, stream>>>(wsf, gamma, beta, wsf + 64, B * T);

  Params P;
  P.x = x; P.bn = wsf + 64;
  P.wih1 = (const float*)d_in[3];  P.whh1 = (const float*)d_in[4];
  P.bih1 = (const float*)d_in[5];  P.bhh1 = (const float*)d_in[6];
  P.wih2 = (const float*)d_in[7];  P.whh2 = (const float*)d_in[8];
  P.bih2 = (const float*)d_in[9];  P.bhh2 = (const float*)d_in[10];
  P.wih3 = (const float*)d_in[11]; P.whh3 = (const float*)d_in[12];
  P.bih3 = (const float*)d_in[13]; P.bhh3 = (const float*)d_in[14];
  P.wih4 = (const float*)d_in[15]; P.whh4 = (const float*)d_in[16];
  P.bih4 = (const float*)d_in[17]; P.bhh4 = (const float*)d_in[18];
  P.fw1 = (const float*)d_in[19]; P.fb1 = (const float*)d_in[20];
  P.fw2 = (const float*)d_in[21]; P.fb2 = (const float*)d_in[22];
  P.fw3 = (const float*)d_in[23]; P.fb3 = (const float*)d_in[24];
  P.fw4 = (const float*)d_in[25]; P.fb4 = (const float*)d_in[26];

  float* out = (float*)d_out;
  size_t BT = (size_t)B * T;
  P.o1 = out; P.o2 = out + BT; P.o3 = out + 4 * BT; P.o4 = out + 7 * BT;

  hipFuncSetAttribute((const void*)ecosys_main,
                      hipFuncAttributeMaxDynamicSharedMemorySize, SMEM_BYTES);
  ecosys_main<<<B / (16 * WAVES), 32 * WAVES, SMEM_BYTES, stream>>>(P);
}
